// IDMPNN_Global_Local_2559800509139
// MI455X (gfx1250) — matrix operand order repaired and checked by reading, not compiled, not run on hardware
//
#include <hip/hip_runtime.h>
#include <hip/hip_bf16.h>
#include <stdint.h>

// ---------------------------------------------------------------------------
// IDMPNN Global/Local — MI455X (gfx1250) implementation.
//
// Dead-code analysis of the reference: the returned value depends ONLY on the
// "z" path (inmlp -> idemb scatter -> local_mlps with sub_g -> set3 -> set4 ->
// segment_max -> out_W). The (NS,NM,PERMDIM,HID) h tensor, adj_g, mlps,
// global_mlps, set1, set2, num_node are all dead. We compute only the live path.
//
// Precision: bf16 WMMA (v_wmma_f32_16x16x32_bf16) with fp32 accumulation for
// the HID x HID GEMMs; LayerNorm / reductions / pooling in fp32.
//
// Data movement: weights are pre-swizzled into WMMA B-fragment order so each
// lane's 16 bf16 values are contiguous (2x ds_load_b128 per fragment); A
// fragments are two contiguous 16B runs per lane; elementwise phases use
// float4 / packed-bf16 vector LDS traffic.
// ---------------------------------------------------------------------------

typedef __bf16 bf16_t;
typedef __bf16 v16bf __attribute__((ext_vector_type(16)));
typedef __bf16 v8bf  __attribute__((ext_vector_type(8)));
typedef __bf16 v4bf  __attribute__((ext_vector_type(4)));
typedef float  v8f   __attribute__((ext_vector_type(8)));

#define HID   64
#define KK    4
#define PDIM  24
#define NB    16
#define NMX   64
#define NSUB  256
#define ZROWS (KK * PDIM)   // 96

// ---------------------------------------------------------------------------
// Input pointer map. Top-level setup_inputs() dict in insertion order; the
// nested 'params' dict flattened as a JAX pytree (dict keys sorted
// alphabetically, lists in order; each mlp dict flattens as W1,W2,b1,b2,be,g):
//  0 x  1 adj  2 subadj  3 subgs(i64)  4 num_subg(i32)  5 num_node(i32)
//  6 edge_W  7 edge_b      8..13 global_mlps[0] (dead)     14 idemb
// 15..20 inmlp: W1 W2 b1 b2 be g     21..26 local_mlps[0]: W1 W2 b1 b2 be g
// 27..38 mlps[0..1] (dead)           39 out_W  40 out_b
// 41..44 set1/set2 (dead)  45 set3.W 46 set3.b 47 set4.W 48 set4.b
// ---------------------------------------------------------------------------

// k-th element (k in 0..3) of the p-th lexicographic permutation of {1,2,3,4}
__device__ inline int perm_digit(int p, int k) {
  int pool[4] = {1, 2, 3, 4};
  const int fact[4] = {6, 2, 1, 1};
  int rem = p;
  for (int i = 0; i < 4; ++i) {
    int d = rem / fact[i];
    rem -= d * fact[i];
    int v = pool[d];
    if (i == k) return v;
    for (int j = d; j < 3; ++j) pool[j] = pool[j + 1];
  }
  return 0;
}

// ---- WMMA fragment helpers (wave32, 16x16x32 bf16) -------------------------
__device__ inline v16bf cat8(v8bf lo, v8bf hi) {
  return __builtin_shufflevector(lo, hi, 0, 1, 2, 3, 4, 5, 6, 7,
                                 8, 9, 10, 11, 12, 13, 14, 15);
}

// A 16x32 (MxK) row-major, ld elements: lanes 0-15 row M=lane hold
// K={0..7,16..23}; lanes 16-31 row M=lane-16 hold K={8..15,24..31}.
// Two contiguous 16B runs per lane -> 2x ds_load_b128.
__device__ inline v16bf load_a16x32(const bf16_t* base, int ld, int lane) {
  const bf16_t* row = base + (lane & 15) * ld + ((lane < 16) ? 0 : 8);
  return cat8(*(const v8bf*)row, *(const v8bf*)(row + 16));
}

// B fragment from weights pre-swizzled into fragment order:
// wswz[(((ks*4+nt)*32)+lane)*16 + e]  (32B contiguous per lane)
__device__ inline v16bf load_b_swz(const bf16_t* wswz, int frag, int lane) {
  const bf16_t* p = wswz + (frag * 32 + lane) * 16;
  return cat8(*(const v8bf*)p, *(const v8bf*)(p + 8));
}

// C/D 16x16 f32: lane = N + 16*(M>=8), vgpr v holds M = v + 8*(lane>=16).
__device__ inline void store_d16x16(float* base, int ld, int lane, v8f d) {
  int n  = lane & 15;
  int mb = (lane < 16) ? 0 : 8;
#pragma unroll
  for (int v = 0; v < 8; ++v) base[(mb + v) * ld + n] = d[v];
}

__device__ inline void atomic_max_f32_nonneg(float* addr, float val) {
  // valid for non-negative floats: IEEE bit pattern is monotone
  atomicMax((unsigned int*)addr, __float_as_uint(val));
}

// one wave computes a 16x64 tile: D = A(16x64) x W(64x64), W pre-swizzled
__device__ inline void gemm_tile_16x64(const bf16_t* Abase, const bf16_t* wswz,
                                       int lane, v8f acc[4]) {
#pragma unroll
  for (int ks = 0; ks < 2; ++ks) {
    v16bf a = load_a16x32(Abase + ks * 32, HID, lane);
#pragma unroll
    for (int nt = 0; nt < 4; ++nt) {
      v16bf b = load_b_swz(wswz, ks * 4 + nt, lane);
      acc[nt] = __builtin_amdgcn_wmma_f32_16x16x32_bf16(
          false, a, false, b, (short)0, acc[nt], false, false);
    }
  }
}

// ---------------------------------------------------------------------------
// Kernel 1: zero pooled buffer, convert + swizzle the three live weight
// matrices to bf16 B-fragment order, derive subgbatch from num_subg.
// ---------------------------------------------------------------------------
__global__ void k_prepare(const float* __restrict__ inW2,
                          const float* __restrict__ lW1,
                          const float* __restrict__ lW2,
                          const int* __restrict__ num_subg,
                          float* __restrict__ pooled,
                          int* __restrict__ subgbatch,
                          bf16_t* __restrict__ inW2b,
                          bf16_t* __restrict__ lW1b,
                          bf16_t* __restrict__ lW2b) {
  const int t = threadIdx.x;
  for (int i = t * 16; i < HID * HID; i += 256 * 16) {   // global_prefetch_b8
    __builtin_prefetch(&inW2[i], 0, 1);
    __builtin_prefetch(&lW1[i], 0, 1);
    __builtin_prefetch(&lW2[i], 0, 1);
  }
  for (int i = t; i < NB * HID; i += 256) pooled[i] = 0.0f;
  for (int i = t; i < HID * HID; i += 256) {
    // i = ((ks*4+nt)*32 + lane)*16 + e  ->  source (k, n)
    int e = i & 15, lane = (i >> 4) & 31, frag = i >> 9;
    int ks = frag >> 2, nt = frag & 3;
    int n = nt * 16 + (lane & 15);
    int k = ks * 32 + ((lane < 16) ? 0 : 8) + (e & 7) + ((e >> 3) << 4);
    int src = k * HID + n;
    inW2b[i] = (bf16_t)inW2[src];
    lW1b[i]  = (bf16_t)lW1[src];
    lW2b[i]  = (bf16_t)lW2[src];
  }
  if (t < NSUB) {  // jnp.repeat(arange(B), num_subg, total_repeat_length=NS)
    int cum = 0, b = NB - 1;
    for (int i = 0; i < NB; ++i) {
      cum += num_subg[i];
      if (t < cum) { b = i; break; }
    }
    subgbatch[t] = b;
  }
}

// row LayerNorm(+bias) + ReLU helper, float4-vectorized over 64 columns
__device__ inline void ln_relu_row(float* o, const float* b2, const float* g,
                                   const float* be, float* out, bool accumulate) {
  float4* ov = (float4*)o;
  const float4* b2v = (const float4*)b2;
  float mu = 0.0f;
#pragma unroll
  for (int c = 0; c < 16; ++c) {
    float4 v = ov[c], bb = b2v[c];
    v.x += bb.x; v.y += bb.y; v.z += bb.z; v.w += bb.w;
    ov[c] = v;
    mu += v.x + v.y + v.z + v.w;
  }
  mu *= (1.0f / HID);
  float var = 0.0f;
#pragma unroll
  for (int c = 0; c < 16; ++c) {
    float4 v = ov[c];
    float dx = v.x - mu, dy = v.y - mu, dz = v.z - mu, dw = v.w - mu;
    var += dx * dx + dy * dy + dz * dz + dw * dw;
  }
  var *= (1.0f / HID);
  const float rs = rsqrtf(var + 1e-5f);
  const float4* gv = (const float4*)g;
  const float4* bev = (const float4*)be;
  float4* outv = (float4*)out;
#pragma unroll
  for (int c = 0; c < 16; ++c) {
    float4 v = ov[c], gg = gv[c], bb = bev[c];
    float4 r;
    r.x = fmaxf((v.x - mu) * rs * gg.x + bb.x, 0.0f);
    r.y = fmaxf((v.y - mu) * rs * gg.y + bb.y, 0.0f);
    r.z = fmaxf((v.z - mu) * rs * gg.z + bb.z, 0.0f);
    r.w = fmaxf((v.w - mu) * rs * gg.w + bb.w, 0.0f);
    if (accumulate) {
      float4 prev = outv[c];
      r.x += prev.x; r.y += prev.y; r.z += prev.z; r.w += prev.w;
    }
    outv[c] = r;
  }
}

// ---------------------------------------------------------------------------
// Kernel 2: h0 = _mlp(x, inmlp)  -> (B*NM=1024, 64) fp32.
// 8 blocks x 256 threads; each wave owns a 16-row tile (8 WMMA per wave).
// ---------------------------------------------------------------------------
__global__ void k_h0(const float* __restrict__ x,  const float* __restrict__ W1,
                     const float* __restrict__ b1, const bf16_t* __restrict__ W2b,
                     const float* __restrict__ b2, const float* __restrict__ g,
                     const float* __restrict__ be, float* __restrict__ h0) {
  extern __shared__ char smem[];
  bf16_t* h1   = (bf16_t*)smem;                    // 128x64 bf16   (16 KB)
  bf16_t* w2s  = (bf16_t*)(smem + 16384);          // swizzled      ( 8 KB)
  float*  outf = (float*)(smem + 16384 + 8192);    // 128x64 f32    (32 KB)
  const int t = threadIdx.x, lane = t & 31, wave = t >> 5;
  const int row0 = blockIdx.x * 128;

  { // stage swizzled W2 with b128 copies
    const uint4* src = (const uint4*)W2b;
    uint4* dst = (uint4*)w2s;
    for (int i = t; i < (HID * HID) / 8; i += 256) dst[i] = src[i];
  }
  { // h1 = relu(x*W1 + b1), float4 math, packed bf16 stores
    const float4* W1v = (const float4*)W1;
    const float4* b1v = (const float4*)b1;
    for (int i = t; i < 128 * 16; i += 256) {
      int r = i >> 4, cg = i & 15;
      float xv = x[row0 + r];
      float4 w = W1v[cg], bb = b1v[cg];
      v4bf o;
      o[0] = (bf16_t)fmaxf(fmaf(xv, w.x, bb.x), 0.0f);
      o[1] = (bf16_t)fmaxf(fmaf(xv, w.y, bb.y), 0.0f);
      o[2] = (bf16_t)fmaxf(fmaf(xv, w.z, bb.z), 0.0f);
      o[3] = (bf16_t)fmaxf(fmaf(xv, w.w, bb.w), 0.0f);
      *(v4bf*)(h1 + r * HID + cg * 4) = o;
    }
  }
  __syncthreads();

  {
    v8f acc[4] = {};
    gemm_tile_16x64(h1 + wave * 16 * HID, w2s, lane, acc);
#pragma unroll
    for (int nt = 0; nt < 4; ++nt)
      store_d16x16(outf + wave * 16 * HID + nt * 16, HID, lane, acc[nt]);
  }
  __syncthreads();

  if (t < 128)
    ln_relu_row(outf + t * HID, b2, g, be, h0 + (size_t)(row0 + t) * HID, false);
}

// ---------------------------------------------------------------------------
// Kernel 3: one block per subgraph (256 blocks x 256 threads = 8 waves).
// Fully fused z pipeline in LDS: z-init -> sub_g einsum -> MLP (2 bf16 WMMA
// GEMMs over 96x64) + LN residual -> perm mean -> set3 -> K-sum -> set4 ->
// atomic segment-max into pooled.
// ---------------------------------------------------------------------------
__global__ void k_z(const float* __restrict__ h0,
                    const long long* __restrict__ subgs,
                    const int* __restrict__ subgbatch,
                    const float* __restrict__ subadj,
                    const float* __restrict__ idemb,
                    const float* __restrict__ edgeW,
                    const float* __restrict__ edgeb,
                    const bf16_t* __restrict__ W1b, const float* __restrict__ b1,
                    const bf16_t* __restrict__ W2b, const float* __restrict__ b2,
                    const float* __restrict__ g,   const float* __restrict__ be,
                    const float* __restrict__ s3W, const float* __restrict__ s3b,
                    const float* __restrict__ s4W, const float* __restrict__ s4b,
                    float* __restrict__ pooled) {
  extern __shared__ char smem[];
  float*  zf  = (float*)smem;                       // 96x64 f32  (24576 B)
  bf16_t* tb  = (bf16_t*)(smem + 24576);            // 96x64 bf16 (12288 B)
  bf16_t* hb  = (bf16_t*)(smem + 36864);            // 96x64 bf16 (12288 B)
  float*  of  = (float*)(smem + 49152);             // 96x64 f32  (24576 B)
  bf16_t* w1s = (bf16_t*)(smem + 73728);            // swizzled   ( 8192 B)
  bf16_t* w2s = (bf16_t*)(smem + 81920);            // swizzled   ( 8192 B)
  float*  zm  = (float*)(smem + 90112);             // 4x64 f32
  float*  zfs = (float*)(smem + 91136);             // 4x64 f32
  float*  zs  = (float*)(smem + 92160);             // 64 f32

  const int s = blockIdx.x;
  const int t = threadIdx.x, lane = t & 31, wave = t >> 5;
  const int bb = subgbatch[s];
  int nd[KK];
#pragma unroll
  for (int k = 0; k < KK; ++k) nd[k] = (int)subgs[s * KK + k];

  { // stage swizzled weights (b128 copies)
    const uint4* s1 = (const uint4*)W1b;
    const uint4* s2 = (const uint4*)W2b;
    uint4* d1 = (uint4*)w1s;
    uint4* d2 = (uint4*)w2s;
    for (int i = t; i < (HID * HID) / 8; i += 256) { d1[i] = s1[i]; d2[i] = s2[i]; }
  }

  // z[k,p,:] = h0[bb, nd[k], :] * idemb[lab(k,p), :]   (float4)
  for (int i = t; i < ZROWS * 16; i += 256) {
    int r = i >> 4, cg = i & 15;
    int k = r / PDIM, p = r - k * PDIM;
    int lab = (k == 0) ? 0 : perm_digit(p, k - 1);
    float4 hv = *(const float4*)(h0 + (size_t)(bb * NMX + nd[k]) * HID + cg * 4);
    float4 iv = *(const float4*)(idemb + lab * HID + cg * 4);
    float4 o;
    o.x = hv.x * iv.x; o.y = hv.y * iv.y; o.z = hv.z * iv.z; o.w = hv.w * iv.w;
    *(float4*)(zf + r * HID + cg * 4) = o;
  }
  __syncthreads();

  // t[i,p,:] = eW*sum_j subadj[i,j] z[j,p,:] + eb*sum_j z[j,p,:]  (float4)
  for (int i = t; i < ZROWS * 16; i += 256) {
    int r = i >> 4, cg = i & 15;
    int ii = r / PDIM, p = r - ii * PDIM;
    float4 s0 = {0, 0, 0, 0}, s1 = {0, 0, 0, 0};
#pragma unroll
    for (int j = 0; j < KK; ++j) {
      float4 zz = *(const float4*)(zf + (j * PDIM + p) * HID + cg * 4);
      float a = subadj[bb * 16 + ii * 4 + j];
      s0.x += zz.x; s0.y += zz.y; s0.z += zz.z; s0.w += zz.w;
      s1.x = fmaf(a, zz.x, s1.x); s1.y = fmaf(a, zz.y, s1.y);
      s1.z = fmaf(a, zz.z, s1.z); s1.w = fmaf(a, zz.w, s1.w);
    }
    float4 ew = *(const float4*)(edgeW + cg * 4);
    float4 eb = *(const float4*)(edgeb + cg * 4);
    v4bf o;
    o[0] = (bf16_t)(ew.x * s1.x + eb.x * s0.x);
    o[1] = (bf16_t)(ew.y * s1.y + eb.y * s0.y);
    o[2] = (bf16_t)(ew.z * s1.z + eb.z * s0.z);
    o[3] = (bf16_t)(ew.w * s1.w + eb.w * s0.w);
    *(v4bf*)(tb + r * HID + cg * 4) = o;
  }
  __syncthreads();

  // GEMM1: hidden = relu(t @ W1 + b1); 96 rows = 6 wave-tiles
  if (wave < 6) {
    v8f acc[4] = {};
    gemm_tile_16x64(tb + wave * 16 * HID, w1s, lane, acc);
    int n0 = lane & 15, mb = (lane < 16) ? 0 : 8;
#pragma unroll
    for (int nt = 0; nt < 4; ++nt)
#pragma unroll
      for (int v = 0; v < 8; ++v) {
        int n = nt * 16 + n0;
        hb[(wave * 16 + mb + v) * HID + n] = (bf16_t)fmaxf(acc[nt][v] + b1[n], 0.0f);
      }
  }
  __syncthreads();

  // GEMM2: o = hidden @ W2
  if (wave < 6) {
    v8f acc[4] = {};
    gemm_tile_16x64(hb + wave * 16 * HID, w2s, lane, acc);
#pragma unroll
    for (int nt = 0; nt < 4; ++nt)
      store_d16x16(of + wave * 16 * HID + nt * 16, HID, lane, acc[nt]);
  }
  __syncthreads();

  // +b2, LayerNorm, ReLU, residual add into z
  if (t < ZROWS)
    ln_relu_row(of + t * HID, b2, g, be, zf + t * HID, true);
  __syncthreads();

  // zmean[k,:] = mean_p z[k,p,:]   (float4; 64 threads)
  if (t < 64) {
    int k = t >> 4, cg = t & 15;
    float4 acc = {0, 0, 0, 0};
    for (int p = 0; p < PDIM; ++p) {
      float4 v = *(const float4*)(zf + (k * PDIM + p) * HID + cg * 4);
      acc.x += v.x; acc.y += v.y; acc.z += v.z; acc.w += v.w;
    }
    acc.x *= (1.0f / PDIM); acc.y *= (1.0f / PDIM);
    acc.z *= (1.0f / PDIM); acc.w *= (1.0f / PDIM);
    *(float4*)(zm + k * HID + cg * 4) = acc;
  }
  __syncthreads();

  // zf[k,c] = relu(zmean @ set3.W + set3.b)
  {
    int k = t >> 6, c = t & 63;
    float acc = s3b[c];
    for (int d = 0; d < HID; ++d) acc = fmaf(zm[k * HID + d], s3W[d * HID + c], acc);
    zfs[t] = fmaxf(acc, 0.0f);
  }
  __syncthreads();

  if (t < HID) {  // sum over K
    float acc = 0.0f;
#pragma unroll
    for (int k = 0; k < KK; ++k) acc += zfs[k * HID + t];
    zs[t] = acc;
  }
  __syncthreads();

  if (t < HID) {  // relu(zsum @ set4.W + set4.b) -> segment max
    float acc = s4b[t];
    for (int d = 0; d < HID; ++d) acc = fmaf(zs[d], s4W[d * HID + t], acc);
    atomic_max_f32_nonneg(&pooled[bb * HID + t], fmaxf(acc, 0.0f));
  }
}

// ---------------------------------------------------------------------------
// Kernel 4: out[b] = pooled[b] @ out_W + out_b   (16 dot products)
// ---------------------------------------------------------------------------
__global__ void k_out(const float* __restrict__ pooled, const float* __restrict__ outW,
                      const float* __restrict__ outb, float* __restrict__ out) {
  int t = threadIdx.x;
  if (t < NB) {
    float acc = outb[0];
    for (int d = 0; d < HID; ++d) acc = fmaf(pooled[t * HID + d], outW[d], acc);
    out[t] = acc;
  }
}

// ---------------------------------------------------------------------------
extern "C" void kernel_launch(void* const* d_in, const int* in_sizes, int n_in,
                              void* d_out, int out_size, void* d_ws, size_t ws_size,
                              hipStream_t stream) {
  (void)in_sizes; (void)n_in; (void)out_size; (void)ws_size;
  const float*     x        = (const float*)d_in[0];
  const float*     subadj   = (const float*)d_in[2];
  const long long* subgs    = (const long long*)d_in[3];
  const int*       num_subg = (const int*)d_in[4];
  const float*     edgeW    = (const float*)d_in[6];
  const float*     edgeb    = (const float*)d_in[7];
  const float*     idemb    = (const float*)d_in[14];
  const float *inW1 = (const float*)d_in[15], *inW2 = (const float*)d_in[16];
  const float *inb1 = (const float*)d_in[17], *inb2 = (const float*)d_in[18];
  const float *inbe = (const float*)d_in[19], *ing  = (const float*)d_in[20];
  const float *lW1  = (const float*)d_in[21], *lW2  = (const float*)d_in[22];
  const float *lb1  = (const float*)d_in[23], *lb2  = (const float*)d_in[24];
  const float *lbe  = (const float*)d_in[25], *lg   = (const float*)d_in[26];
  const float *outW = (const float*)d_in[39], *outb = (const float*)d_in[40];
  const float *s3W  = (const float*)d_in[45], *s3b  = (const float*)d_in[46];
  const float *s4W  = (const float*)d_in[47], *s4b  = (const float*)d_in[48];

  char* ws = (char*)d_ws;                       // ~289 KB used
  float*  pooled    = (float*)(ws + 0);         // 16*64 f32
  int*    subgbatch = (int*)(ws + 4096);        // 256 i32
  float*  h0        = (float*)(ws + 8192);      // 1024*64 f32
  bf16_t* inW2b     = (bf16_t*)(ws + 8192 + 262144);   // swizzled bf16 weights
  bf16_t* lW1b      = inW2b + HID * HID;
  bf16_t* lW2b      = lW1b + HID * HID;

  hipLaunchKernelGGL(k_prepare, dim3(1), dim3(256), 0, stream,
                     inW2, lW1, lW2, num_subg, pooled, subgbatch, inW2b, lW1b, lW2b);
  hipLaunchKernelGGL(k_h0, dim3(8), dim3(256), 57344, stream,
                     x, inW1, inb1, inW2b, inb2, ing, inbe, h0);
  hipLaunchKernelGGL(k_z, dim3(NSUB), dim3(256), 92480, stream,
                     h0, subgs, subgbatch, subadj, idemb, edgeW, edgeb,
                     lW1b, lb1, lW2b, lb2, lg, lbe, s3W, s3b, s4W, s4b, pooled);
  hipLaunchKernelGGL(k_out, dim3(1), dim3(32), 0, stream,
                     pooled, outW, outb, (float*)d_out);
}
